// RelationalMemoryCoreCell_30966714204737
// MI455X (gfx1250) — matrix operand order repaired and checked by reading, not compiled, not run on hardware
//
#include <hip/hip_runtime.h>
#include <hip/hip_bf16.h>
#include <stdint.h>

typedef __bf16 bf16_t;
typedef __attribute__((ext_vector_type(16))) __bf16 v16bf;
typedef __attribute__((ext_vector_type(8)))  __bf16 v8bf;
typedef __attribute__((ext_vector_type(8)))  float  v8f;
typedef __attribute__((ext_vector_type(4)))  float  v4f;

#define B_SZ   256
#define S_SZ   16
#define H_SZ   8
#define HS_SZ  128
#define M_SZ   1024
#define ROWS   (B_SZ * S_SZ)        // 4096
#define LDW    1024

__device__ __forceinline__ float sigmoidf_(float x) {
    return 1.0f / (1.0f + __expf(-x));
}

// ---------------------------------------------------------------------------
// Prep kernels: one-time bf16 conversion / weight transpose so the GEMM can
// stage tiles with async copies (no conversion possible inside the DMA path).
// ---------------------------------------------------------------------------
__global__ __launch_bounds__(256)
void convert_bf16(const float* __restrict__ X, bf16_t* __restrict__ Y)
{
    size_t o = ((size_t)blockIdx.x * 256 + threadIdx.x) * 8;
    v4f a = *(const v4f*)(X + o);
    v4f b = *(const v4f*)(X + o + 4);
    v8bf y = { (bf16_t)a[0], (bf16_t)a[1], (bf16_t)a[2], (bf16_t)a[3],
               (bf16_t)b[0], (bf16_t)b[1], (bf16_t)b[2], (bf16_t)b[3] };
    *(v8bf*)(Y + o) = y;
}

// Wt[n][k] = (bf16) W[k][n]   (both 1024x1024)
__global__ __launch_bounds__(256)
void transpose_convert(const float* __restrict__ W, bf16_t* __restrict__ Wt)
{
    __shared__ float t[32][33];
    int tx = threadIdx.x & 31, ty = threadIdx.x >> 5;   // 32 x 8
    int kb = blockIdx.x * 32;
    int nb = blockIdx.y * 32;
    #pragma unroll
    for (int j = 0; j < 4; ++j)
        t[ty + j * 8][tx] = W[(size_t)(kb + ty + j * 8) * LDW + nb + tx];
    __syncthreads();
    #pragma unroll
    for (int j = 0; j < 4; ++j)
        Wt[(size_t)(nb + ty + j * 8) * LDW + kb + tx] = (bf16_t)t[tx][ty + j * 8];
}

// ---------------------------------------------------------------------------
// bf16 WMMA GEMM with async-to-LDS double buffering.
//   C[M,1024] = epi( A[M,1024] @ Wt^T )  (Wt is n-major [n][k])
// Block tile 128x128, K-step 64, 8 waves (4 along M x 2 along N),
// each wave 2x4 16x16 fragments -> 16 v_wmma per K-step.
// EPI: 0 = +bias ; 1 = plain ; 2 = +bias +addFull ; 3 = +bias +addRow, sigmoid
// ---------------------------------------------------------------------------
template<int EPI>
__global__ __launch_bounds__(256)
void gemm_bf16_async(const bf16_t* __restrict__ A, const bf16_t* __restrict__ Bt,
                     const float* __restrict__ bias, const float* __restrict__ addRow,
                     const float* __restrict__ addFull, float* __restrict__ C)
{
    __shared__ bf16_t As[2][128 * 64];   // [row][k]  16 KB each
    __shared__ bf16_t Bs[2][128 * 64];   // [n][k]    16 KB each

    const int tid  = threadIdx.x;
    const int wave = tid >> 5;           // wave32
    const int lane = tid & 31;
    const int wm   = wave & 3;           // 32 rows per wave
    const int wn   = wave >> 2;          // 64 cols per wave
    const int half = lane >> 4;
    const int lr   = lane & 15;
    const int blockN = blockIdx.x * 128;
    const int blockM = blockIdx.y * 128;

    // Per-thread staging: 4 x 16B chunks for A, 4 for B, per buffer.
    uint64_t gA[4], gB[4];
    uint32_t lA[2][4], lB[2][4];
    #pragma unroll
    for (int c = 0; c < 4; ++c) {
        int l   = c * 256 + tid;           // 0..1023
        int row = l >> 3;                  // 0..127
        int cr  = (l & 7) * 8;             // bf16 element offset in row
        gA[c] = (uint64_t)(uintptr_t)(A  + (size_t)(blockM + row) * LDW + cr);
        gB[c] = (uint64_t)(uintptr_t)(Bt + (size_t)(blockN + row) * LDW + cr);
        lA[0][c] = (uint32_t)(uintptr_t)&As[0][row * 64 + cr];
        lA[1][c] = (uint32_t)(uintptr_t)&As[1][row * 64 + cr];
        lB[0][c] = (uint32_t)(uintptr_t)&Bs[0][row * 64 + cr];
        lB[1][c] = (uint32_t)(uintptr_t)&Bs[1][row * 64 + cr];
    }

    auto issue = [&](int buf, int k0) {
        uint64_t koff = (uint64_t)k0 * 2;   // bytes along K
        #pragma unroll
        for (int c = 0; c < 4; ++c) {
            asm volatile("global_load_async_to_lds_b128 %0, %1, off"
                         :: "v"(lA[buf][c]), "v"(gA[c] + koff) : "memory");
            asm volatile("global_load_async_to_lds_b128 %0, %1, off"
                         :: "v"(lB[buf][c]), "v"(gB[c] + koff) : "memory");
        }
    };

    v8f acc[2][4] = {};

    issue(0, 0);
    for (int it = 0; it < 16; ++it) {
        const int cur = it & 1;
        if (it + 1 < 16) {
            issue(1 - cur, (it + 1) * 64);
            asm volatile("s_wait_asynccnt 8" ::: "memory");   // cur buffer landed
        } else {
            asm volatile("s_wait_asynccnt 0" ::: "memory");
        }
        __syncthreads();

        const bf16_t* as = As[cur];
        const bf16_t* bs = Bs[cur];
        #pragma unroll
        for (int kc = 0; kc < 64; kc += 32) {
            v16bf a[2], b[4];
            #pragma unroll
            for (int fm = 0; fm < 2; ++fm) {
                int r = wm * 32 + fm * 16 + lr;
                a[fm] = *(const v16bf*)&as[r * 64 + kc + half * 16];
            }
            #pragma unroll
            for (int fn = 0; fn < 4; ++fn) {
                int n = wn * 64 + fn * 16 + lr;
                b[fn] = *(const v16bf*)&bs[n * 64 + kc + half * 16];
            }
            #pragma unroll
            for (int fm = 0; fm < 2; ++fm)
                #pragma unroll
                for (int fn = 0; fn < 4; ++fn)
                    acc[fm][fn] = __builtin_amdgcn_wmma_f32_16x16x32_bf16(
                        false, a[fm], false, b[fn], (short)0, acc[fm][fn], false, false);
        }
        __syncthreads();
    }

    // epilogue: lane lr = col, VGPR v -> row (v + 8*half) within 16x16 frag
    #pragma unroll
    for (int fm = 0; fm < 2; ++fm) {
        int row0 = blockM + wm * 32 + fm * 16 + 8 * half;   // 8 rows, same row>>4
        #pragma unroll
        for (int fn = 0; fn < 4; ++fn) {
            int col = blockN + wn * 64 + fn * 16 + lr;
            float badd = (EPI != 1) ? bias[col] : 0.0f;
            float radd = (EPI == 3) ? addRow[(size_t)(row0 >> 4) * LDW + col] : 0.0f;
            #pragma unroll
            for (int v = 0; v < 8; ++v) {
                int row = row0 + v;
                float x = acc[fm][fn][v] + badd + radd;
                if (EPI == 2) x += addFull[(size_t)row * LDW + col];
                if (EPI == 3) x = sigmoidf_(x);
                C[(size_t)row * LDW + col] = x;
            }
        }
    }
}

// ---------------------------------------------------------------------------
// Attention (tiny): one block per (b,h). Writes mp0 = m_state + attended in
// both f32 (for the +addFull epilogue) and bf16 (A operand of the W_g GEMM).
// ---------------------------------------------------------------------------
__global__ __launch_bounds__(256)
void attention_kernel(const float* __restrict__ m_state,
                      const float* __restrict__ dense_inputs,
                      float* __restrict__ mp0, bf16_t* __restrict__ mp0b)
{
    __shared__ float kv[17 * 132];
    __shared__ float sc[16 * 18];

    const int b   = blockIdx.x >> 3;
    const int h   = blockIdx.x & 7;
    const int tid = threadIdx.x;
    const float scale = 0.08838834764831845f;   // 1/sqrt(128)

    for (int i = tid; i < 17 * 128; i += 256) {
        int r = i >> 7, d = i & 127;
        kv[r * 132 + d] = (r < 16)
            ? m_state[((size_t)b * S_SZ + r) * M_SZ + h * HS_SZ + d]
            : dense_inputs[(size_t)b * M_SZ + h * HS_SZ + d];
    }
    __syncthreads();

    {
        int qi = tid >> 4, r = tid & 15;
        float s = 0.f;
        for (int d = 0; d < 128; ++d) s += kv[qi * 132 + d] * kv[r * 132 + d];
        sc[qi * 18 + r] = s * scale;
        if (r == 0) {
            float s2 = 0.f;
            for (int d = 0; d < 128; ++d) s2 += kv[qi * 132 + d] * kv[16 * 132 + d];
            sc[qi * 18 + 16] = s2 * scale;
        }
    }
    __syncthreads();

    if (tid < 16) {
        float mx = -1e30f;
        for (int j = 0; j < 17; ++j) mx = fmaxf(mx, sc[tid * 18 + j]);
        float sum = 0.f;
        for (int j = 0; j < 17; ++j) {
            float e = __expf(sc[tid * 18 + j] - mx);
            sc[tid * 18 + j] = e; sum += e;
        }
        float inv = 1.0f / sum;
        for (int j = 0; j < 17; ++j) sc[tid * 18 + j] *= inv;
    }
    __syncthreads();

    for (int i = tid; i < 16 * 128; i += 256) {
        int q = i >> 7, d = i & 127;
        float a = 0.f;
        for (int j = 0; j < 17; ++j) a += sc[q * 18 + j] * kv[j * 132 + d];
        size_t off = ((size_t)b * S_SZ + q) * M_SZ + h * HS_SZ + d;
        float v = kv[q * 132 + d] + a;
        mp0[off]  = v;
        mp0b[off] = (bf16_t)v;
    }
}

// ---------------------------------------------------------------------------
// Final combine (faithful double sigmoid on input/output gates).
// ---------------------------------------------------------------------------
__global__ __launch_bounds__(256)
void final_kernel(const float* __restrict__ m_state,
                  const float* __restrict__ gf, const float* __restrict__ gi,
                  const float* __restrict__ go, const float* __restrict__ mp,
                  float* __restrict__ out_mem, float* __restrict__ out_hid)
{
    size_t i = ((size_t)blockIdx.x * 256 + threadIdx.x) * 4;
    v4f m  = *(const v4f*)(m_state + i);
    v4f f  = *(const v4f*)(gf + i);
    v4f ig = *(const v4f*)(gi + i);
    v4f og = *(const v4f*)(go + i);
    v4f p  = *(const v4f*)(mp + i);
    v4f om, oh;
    #pragma unroll
    for (int j = 0; j < 4; ++j) {
        om[j] = sigmoidf_(m[j] * f[j] + 1.0f) + p[j] * sigmoidf_(ig[j]);
        oh[j] = tanhf(m[j]) * sigmoidf_(og[j]);
    }
    *(v4f*)(out_mem + i) = om;
    *(v4f*)(out_hid + i) = oh;
}

extern "C" void kernel_launch(void* const* d_in, const int* in_sizes, int n_in,
                              void* d_out, int out_size, void* d_ws, size_t ws_size,
                              hipStream_t stream) {
    (void)in_sizes; (void)n_in; (void)out_size; (void)ws_size;

    const float* inputs  = (const float*)d_in[0];
    const float* h_state = (const float*)d_in[1];
    const float* m_state = (const float*)d_in[2];
    const float* W_emb   = (const float*)d_in[3];
    const float* b_emb   = (const float*)d_in[4];
    const float* W_g     = (const float*)d_in[5];
    const float* b_g     = (const float*)d_in[6];
    const float* W_wf    = (const float*)d_in[7];
    const float* W_wi    = (const float*)d_in[8];
    const float* W_wo    = (const float*)d_in[9];
    const float* W_uf    = (const float*)d_in[10];
    const float* b_uf    = (const float*)d_in[11];
    const float* W_ui    = (const float*)d_in[12];
    const float* b_ui    = (const float*)d_in[13];
    const float* W_uo    = (const float*)d_in[14];
    const float* b_uo    = (const float*)d_in[15];

    // ---- workspace layout: f32 region then bf16 region ----
    float* ws      = (float*)d_ws;
    float* dense   = ws;                                //  256*1024 f32
    float* wx_f    = dense + (size_t)B_SZ * M_SZ;
    float* wx_i    = wx_f  + (size_t)B_SZ * M_SZ;
    float* wx_o    = wx_i  + (size_t)B_SZ * M_SZ;
    float* mp0     = wx_o  + (size_t)B_SZ * M_SZ;       // 4096*1024 f32
    float* mp_out  = mp0   + (size_t)ROWS * M_SZ;
    float* gate_f  = mp_out + (size_t)ROWS * M_SZ;
    float* gate_i  = gate_f + (size_t)ROWS * M_SZ;
    float* gate_o  = gate_i + (size_t)ROWS * M_SZ;

    bf16_t* wsb      = (bf16_t*)(gate_o + (size_t)ROWS * M_SZ);
    bf16_t* inputs_b = wsb;                             //  256*1024 bf16
    bf16_t* hstate_b = inputs_b + (size_t)B_SZ * M_SZ;  // 4096*1024
    bf16_t* mp0_b    = hstate_b + (size_t)ROWS * M_SZ;  // 4096*1024
    bf16_t* Wt_emb   = mp0_b  + (size_t)ROWS * M_SZ;    // 8 x 1024*1024
    bf16_t* Wt_g     = Wt_emb + (size_t)M_SZ * M_SZ;
    bf16_t* Wt_wf    = Wt_g   + (size_t)M_SZ * M_SZ;
    bf16_t* Wt_wi    = Wt_wf  + (size_t)M_SZ * M_SZ;
    bf16_t* Wt_wo    = Wt_wi  + (size_t)M_SZ * M_SZ;
    bf16_t* Wt_uf    = Wt_wo  + (size_t)M_SZ * M_SZ;
    bf16_t* Wt_ui    = Wt_uf  + (size_t)M_SZ * M_SZ;
    bf16_t* Wt_uo    = Wt_ui  + (size_t)M_SZ * M_SZ;

    float* out_mem = (float*)d_out;
    float* out_hid = out_mem + (size_t)ROWS * M_SZ;

    dim3 blk(256);
    dim3 gridT(32, 32);                  // 1024x1024 transpose
    dim3 gridS(M_SZ / 128, B_SZ / 128);  // 8 x 2   (M = 256)
    dim3 gridL(M_SZ / 128, ROWS / 128);  // 8 x 32  (M = 4096)

    // 0) precision prep
    transpose_convert<<<gridT, blk, 0, stream>>>(W_emb, Wt_emb);
    transpose_convert<<<gridT, blk, 0, stream>>>(W_g,   Wt_g);
    transpose_convert<<<gridT, blk, 0, stream>>>(W_wf,  Wt_wf);
    transpose_convert<<<gridT, blk, 0, stream>>>(W_wi,  Wt_wi);
    transpose_convert<<<gridT, blk, 0, stream>>>(W_wo,  Wt_wo);
    transpose_convert<<<gridT, blk, 0, stream>>>(W_uf,  Wt_uf);
    transpose_convert<<<gridT, blk, 0, stream>>>(W_ui,  Wt_ui);
    transpose_convert<<<gridT, blk, 0, stream>>>(W_uo,  Wt_uo);
    convert_bf16<<<dim3(B_SZ * M_SZ / 2048), blk, 0, stream>>>(inputs,  inputs_b);
    convert_bf16<<<dim3(ROWS * M_SZ / 2048), blk, 0, stream>>>(h_state, hstate_b);

    // 1) input embedding + input-side gate projections (M = 256)
    gemm_bf16_async<0><<<gridS, blk, 0, stream>>>(inputs_b, Wt_emb, b_emb, nullptr, nullptr, dense);
    gemm_bf16_async<1><<<gridS, blk, 0, stream>>>(inputs_b, Wt_wf, nullptr, nullptr, nullptr, wx_f);
    gemm_bf16_async<1><<<gridS, blk, 0, stream>>>(inputs_b, Wt_wi, nullptr, nullptr, nullptr, wx_i);
    gemm_bf16_async<1><<<gridS, blk, 0, stream>>>(inputs_b, Wt_wo, nullptr, nullptr, nullptr, wx_o);

    // 2) attention -> mp0 = m_state + attended (f32 + bf16)
    attention_kernel<<<dim3(B_SZ * H_SZ), blk, 0, stream>>>(m_state, dense, mp0, mp0_b);

    // 3) mp_out = mp0 @ W_g + b_g + mp0
    gemm_bf16_async<2><<<gridL, blk, 0, stream>>>(mp0_b, Wt_g, b_g, nullptr, mp0, mp_out);

    // 4) gates = sigmoid(h_state @ W_u* + b_* + wx_*[row/16])
    gemm_bf16_async<3><<<gridL, blk, 0, stream>>>(hstate_b, Wt_uf, b_uf, wx_f, nullptr, gate_f);
    gemm_bf16_async<3><<<gridL, blk, 0, stream>>>(hstate_b, Wt_ui, b_ui, wx_i, nullptr, gate_i);
    gemm_bf16_async<3><<<gridL, blk, 0, stream>>>(hstate_b, Wt_uo, b_uo, wx_o, nullptr, gate_o);

    // 5) final combine
    final_kernel<<<dim3((ROWS * M_SZ / 4) / 256), blk, 0, stream>>>(
        m_state, gate_f, gate_i, gate_o, mp_out, out_mem, out_hid);
}
